// StableHierarchicalPooling_56753697849444
// MI455X (gfx1250) — matrix operand order repaired and checked by reading, not compiled, not run on hardware
//
#include <hip/hip_runtime.h>
#include <hip/hip_bf16.h>

typedef __attribute__((ext_vector_type(16))) _Float16 v16h;
typedef __attribute__((ext_vector_type(8)))  _Float16 v8h;
typedef __attribute__((ext_vector_type(8)))  float    v8f;

#define N_NODES 50000
#define C_DIM   128
#define K_SUP   32
#define B_GRAPHS 16
#define EPSV    1e-9f

// ---- workspace layout (floats) ----
#define AVG_OFF  0      // [32]  sum_n s[n,k]
#define ENT_OFF  32     // [1]   sum_n sum_k s*log(s+eps)
#define MU_OFF   64     // [32*2] sum_n s*pos
#define P_OFF    128    // [32]  sum_n s*|pos|^2
#define BPOS_OFF 160    // [16*32*2] per-graph sum s*pos
#define BSUM_OFF 1184   // [16*32]   per-graph sum s
#define WS_FLOATS 1696

// ---- d_out layout (floats) ----
#define OUT_CNT  (B_GRAPHS * K_SUP * C_DIM)   // 65536
#define S_OFF    OUT_CNT
#define S_CNT    (N_NODES * K_SUP)            // 1600000
#define MUOUT_OFF (S_OFF + S_CNT)             // 1665536
#define LOSS_OFF  (MUOUT_OFF + B_GRAPHS * K_SUP * 2) // 1666560

// ---- dynamic LDS layout (bytes), all 16B aligned ----
#define LDS_W1T   0        // [128 out][128 in] half = 32768
#define LDS_W2T   32768    // [32 out][128 in] half  = 8192
#define LDS_B1    40960    // 128 f32      = 512
#define LDS_B2    41472    // 32 f32       = 128
#define LDS_MASK  41600    // 32 f32       = 128
#define LDS_XHT   41728    // [128 ch][64 row] half = 16384
#define LDS_HH    58112    // 4 waves x [16 row][128 k] half = 16384
#define LDS_SRAW  74496    // [64 row][32 k] f32 = 8192
#define LDS_SMKT  82688    // [32 k][64 row] half = 4096
#define LDS_BATCH 86784    // 64 int = 256
#define LDS_TOTAL 87040

__device__ __forceinline__ float gumbel_noise(unsigned idx) {
  // fixed-seed hash -> uniform (0,1) -> gumbel (stand-in for jax threefry, key=42)
  unsigned h = idx * 0x9E3779B1u + 42u;
  h ^= h >> 16; h *= 0x85EBCA6Bu;
  h ^= h >> 13; h *= 0xC2B2AE35u;
  h ^= h >> 16;
  float u = ((float)(h >> 9) + 0.5f) * (1.0f / 8388608.0f);
  return -__logf(-__logf(u));
}

__device__ __forceinline__ float wave_sum32(float v) {
  #pragma unroll
  for (int o = 16; o > 0; o >>= 1) v += __shfl_xor(v, o, 32);
  return v;
}

__device__ __forceinline__ v16h cat16(v8h a, v8h b) {
  return __builtin_shufflevector(a, b, 0, 1, 2, 3, 4, 5, 6, 7,
                                        8, 9, 10, 11, 12, 13, 14, 15);
}

__global__ __launch_bounds__(256) void shp_zero_kernel(float* out_head, float* ws) {
  int i = blockIdx.x * blockDim.x + threadIdx.x;
  int stride = gridDim.x * blockDim.x;
  for (int t = i; t < OUT_CNT; t += stride) out_head[t] = 0.0f;
  for (int t = i; t < WS_FLOATS; t += stride) ws[t] = 0.0f;
}

__global__ __launch_bounds__(128) void shp_main_kernel(
    const float* __restrict__ x, const float* __restrict__ pos,
    const float* __restrict__ W1, const float* __restrict__ b1,
    const float* __restrict__ W2, const float* __restrict__ b2,
    const float* __restrict__ scaling, const float* __restrict__ amask,
    const int* __restrict__ batch,
    float* __restrict__ d_pool, float* __restrict__ d_s, float* __restrict__ ws) {
  extern __shared__ char smem[];
  _Float16* w1t   = (_Float16*)(smem + LDS_W1T);   // [out col][in k]
  _Float16* w2t   = (_Float16*)(smem + LDS_W2T);   // [out col][in k]
  float*    b1f   = (float*)(smem + LDS_B1);
  float*    b2f   = (float*)(smem + LDS_B2);
  float*    maskf = (float*)(smem + LDS_MASK);
  _Float16* xhT   = (_Float16*)(smem + LDS_XHT);   // [channel][local row]
  _Float16* hhall = (_Float16*)(smem + LDS_HH);    // per-wave [row][k]
  float*    sraw  = (float*)(smem + LDS_SRAW);     // [local row][k]
  _Float16* smkT  = (_Float16*)(smem + LDS_SMKT);  // [k_super][local row]
  int*      batL  = (int*)(smem + LDS_BATCH);

  const int tid  = threadIdx.x;
  const int wave = tid >> 5;
  const int lane = tid & 31;
  const int l15  = lane & 15;
  const int hi   = lane >> 4;
  const int rbase = blockIdx.x * 64;            // first row of block
  const int r0    = rbase + wave * 16;          // first row of this wave's tile

  // ---- stage weights transposed (f32 -> f16), biases, batch ----
  for (int i = tid * 4; i < C_DIM * C_DIM; i += 128 * 4) {
    float4 w = *(const float4*)(W1 + i);        // row k = i>>7, cols c..c+3
    int k = i >> 7, c = i & 127;
    w1t[(c + 0) * C_DIM + k] = (_Float16)w.x;
    w1t[(c + 1) * C_DIM + k] = (_Float16)w.y;
    w1t[(c + 2) * C_DIM + k] = (_Float16)w.z;
    w1t[(c + 3) * C_DIM + k] = (_Float16)w.w;
  }
  for (int i = tid * 4; i < C_DIM * K_SUP; i += 128 * 4) {
    float4 w = *(const float4*)(W2 + i);        // row k = i>>5, cols c..c+3
    int k = i >> 5, c = i & 31;
    w2t[(c + 0) * C_DIM + k] = (_Float16)w.x;
    w2t[(c + 1) * C_DIM + k] = (_Float16)w.y;
    w2t[(c + 2) * C_DIM + k] = (_Float16)w.z;
    w2t[(c + 3) * C_DIM + k] = (_Float16)w.w;
  }
  if (tid < C_DIM) b1f[tid] = b1[tid];
  if (tid < K_SUP) { b2f[tid] = b2[tid]; maskf[tid] = amask[tid]; }
  if (tid < 64) {
    int g = rbase + tid;
    batL[tid] = (g < N_NODES) ? batch[g] : -1;
  }
  __syncthreads();
  const int b0 = batL[0];

  // ---- load x tile (vectorized) into A-fragments + transposed mirror xhT ----
  v16h afrag[4];
  {
    int grow = r0 + l15;
    int rr = (grow < N_NODES) ? grow : (N_NODES - 1);
    int lrow = wave * 16 + l15;
    const float4* xrow = (const float4*)(x + (size_t)rr * C_DIM);
    __builtin_prefetch(&x[(size_t)((rr + 64 < N_NODES) ? rr + 64 : rr) * C_DIM], 0, 0);
    #pragma unroll
    for (int kk = 0; kk < 4; ++kk) {
      int kb0 = kk * 32 + hi * 8;        // halves j=0..7  -> k = kb0 + j
      int kb1 = kk * 32 + 16 + hi * 8;   // halves j=8..15 -> k = kb1 + (j-8)
      float4 q0 = xrow[kb0 >> 2];
      float4 q1 = xrow[(kb0 >> 2) + 1];
      float4 q2 = xrow[kb1 >> 2];
      float4 q3 = xrow[(kb1 >> 2) + 1];
      float vals[16] = {q0.x, q0.y, q0.z, q0.w, q1.x, q1.y, q1.z, q1.w,
                        q2.x, q2.y, q2.z, q2.w, q3.x, q3.y, q3.z, q3.w};
      #pragma unroll
      for (int j = 0; j < 16; ++j) {
        int k = (j < 8) ? (kb0 + j) : (kb1 + j - 8);
        _Float16 hv = (_Float16)vals[j];
        afrag[kk][j] = hv;
        xhT[k * 64 + lrow] = hv;
      }
    }
  }

  // ---- GEMM1: h = relu(x @ W1 + b1); stage h per-wave row-major ----
  _Float16* hh = hhall + wave * 16 * C_DIM;
  #pragma unroll
  for (int jb = 0; jb < 8; ++jb) {
    int col = jb * 16 + l15;
    v8f c = {};
    #pragma unroll
    for (int kk = 0; kk < 4; ++kk) {
      const _Float16* bp = w1t + col * C_DIM + kk * 32 + hi * 8;
      v16h bw = cat16(*(const v8h*)bp, *(const v8h*)(bp + 16));
      c = __builtin_amdgcn_wmma_f32_16x16x32_f16(false, afrag[kk], false, bw,
                                                 (short)0, c, false, false);
    }
    float bias = b1f[col];
    #pragma unroll
    for (int v = 0; v < 8; ++v) {
      float hv = fmaxf(c[v] + bias, 0.0f);
      hh[(v + 8 * hi) * C_DIM + col] = (_Float16)hv;
    }
  }

  // ---- rebuild h as A-fragments (wide loads; DS in-order per wave) ----
  v16h hfrag[4];
  #pragma unroll
  for (int kk = 0; kk < 4; ++kk) {
    const _Float16* hp = hh + l15 * C_DIM + kk * 32 + hi * 8;
    hfrag[kk] = cat16(*(const v8h*)hp, *(const v8h*)(hp + 16));
  }

  // ---- GEMM2: logits = (h @ W2 + b2)*scaling, mask, gumbel ----
  const float sc = scaling[0];
  float z0[8], z1[8];
  #pragma unroll
  for (int j2 = 0; j2 < 2; ++j2) {
    int col = j2 * 16 + l15;
    v8f c = {};
    #pragma unroll
    for (int kk = 0; kk < 4; ++kk) {
      const _Float16* bp = w2t + col * C_DIM + kk * 32 + hi * 8;
      v16h bw = cat16(*(const v8h*)bp, *(const v8h*)(bp + 16));
      c = __builtin_amdgcn_wmma_f32_16x16x32_f16(false, hfrag[kk], false, bw,
                                                 (short)0, c, false, false);
    }
    float bias = b2f[col];
    float mk = maskf[col];
    #pragma unroll
    for (int v = 0; v < 8; ++v) {
      int grow = r0 + v + 8 * hi;
      float lg = (c[v] + bias) * sc;
      if (mk == 0.0f) lg = -1e9f;
      float g = gumbel_noise((unsigned)(grow * K_SUP + col));
      float zz = lg + g;  // TAU = 1.0
      if (j2 == 0) z0[v] = zz; else z1[v] = zz;
    }
  }

  // ---- row-wise softmax over K=32 (16-lane xor reductions) ----
  float s0[8], s1[8];
  #pragma unroll
  for (int v = 0; v < 8; ++v) {
    float m = fmaxf(z0[v], z1[v]);
    #pragma unroll
    for (int o = 8; o > 0; o >>= 1) m = fmaxf(m, __shfl_xor(m, o, 32));
    float e0 = __expf(z0[v] - m), e1 = __expf(z1[v] - m);
    float sm = e0 + e1;
    #pragma unroll
    for (int o = 8; o > 0; o >>= 1) sm += __shfl_xor(sm, o, 32);
    s0[v] = e0 / sm;
    s1[v] = e1 / sm;
  }

  // ---- emit s, stage sraw/smkT, accumulate loss partials ----
  const int col0 = l15, col1 = 16 + l15;
  float ent_part = 0.0f;
  float a0 = 0, a1 = 0, mx0 = 0, my0 = 0, mx1 = 0, my1 = 0, p0 = 0, p1 = 0;
  float bpx0 = 0, bpy0 = 0, bs0 = 0, bpx1 = 0, bpy1 = 0, bs1 = 0;
  #pragma unroll
  for (int v = 0; v < 8; ++v) {
    int grow = r0 + v + 8 * hi;
    int lr = wave * 16 + v + 8 * hi;
    bool valid = grow < N_NODES;
    float sv0 = valid ? s0[v] : 0.0f;
    float sv1 = valid ? s1[v] : 0.0f;
    sraw[lr * K_SUP + col0] = sv0;
    sraw[lr * K_SUP + col1] = sv1;
    int bb = batL[lr];
    bool inb0 = valid && (bb == b0);
    smkT[col0 * 64 + lr] = (_Float16)(inb0 ? sv0 : 0.0f);
    smkT[col1 * 64 + lr] = (_Float16)(inb0 ? sv1 : 0.0f);
    if (valid) {
      d_s[(size_t)grow * K_SUP + col0] = sv0;
      d_s[(size_t)grow * K_SUP + col1] = sv1;
      float px = pos[grow * 2 + 0], py = pos[grow * 2 + 1];
      float psq = px * px + py * py;
      ent_part += sv0 * __logf(sv0 + EPSV) + sv1 * __logf(sv1 + EPSV);
      a0 += sv0; a1 += sv1;
      mx0 += sv0 * px; my0 += sv0 * py; mx1 += sv1 * px; my1 += sv1 * py;
      p0 += sv0 * psq; p1 += sv1 * psq;
      if (inb0) {
        bpx0 += sv0 * px; bpy0 += sv0 * py; bs0 += sv0;
        bpx1 += sv1 * px; bpy1 += sv1 * py; bs1 += sv1;
      } else {
        atomicAdd(&ws[BPOS_OFF + (bb * K_SUP + col0) * 2 + 0], sv0 * px);
        atomicAdd(&ws[BPOS_OFF + (bb * K_SUP + col0) * 2 + 1], sv0 * py);
        atomicAdd(&ws[BSUM_OFF + bb * K_SUP + col0], sv0);
        atomicAdd(&ws[BPOS_OFF + (bb * K_SUP + col1) * 2 + 0], sv1 * px);
        atomicAdd(&ws[BPOS_OFF + (bb * K_SUP + col1) * 2 + 1], sv1 * py);
        atomicAdd(&ws[BSUM_OFF + bb * K_SUP + col1], sv1);
      }
    }
  }
  ent_part = wave_sum32(ent_part);
  if (lane == 0) atomicAdd(&ws[ENT_OFF], ent_part);
  a0 += __shfl_xor(a0, 16, 32);   a1 += __shfl_xor(a1, 16, 32);
  mx0 += __shfl_xor(mx0, 16, 32); my0 += __shfl_xor(my0, 16, 32);
  mx1 += __shfl_xor(mx1, 16, 32); my1 += __shfl_xor(my1, 16, 32);
  p0 += __shfl_xor(p0, 16, 32);   p1 += __shfl_xor(p1, 16, 32);
  bpx0 += __shfl_xor(bpx0, 16, 32); bpy0 += __shfl_xor(bpy0, 16, 32);
  bs0 += __shfl_xor(bs0, 16, 32);
  bpx1 += __shfl_xor(bpx1, 16, 32); bpy1 += __shfl_xor(bpy1, 16, 32);
  bs1 += __shfl_xor(bs1, 16, 32);
  if (hi == 0) {
    atomicAdd(&ws[AVG_OFF + col0], a0);  atomicAdd(&ws[AVG_OFF + col1], a1);
    atomicAdd(&ws[MU_OFF + col0 * 2 + 0], mx0); atomicAdd(&ws[MU_OFF + col0 * 2 + 1], my0);
    atomicAdd(&ws[MU_OFF + col1 * 2 + 0], mx1); atomicAdd(&ws[MU_OFF + col1 * 2 + 1], my1);
    atomicAdd(&ws[P_OFF + col0], p0);    atomicAdd(&ws[P_OFF + col1], p1);
    atomicAdd(&ws[BPOS_OFF + (b0 * K_SUP + col0) * 2 + 0], bpx0);
    atomicAdd(&ws[BPOS_OFF + (b0 * K_SUP + col0) * 2 + 1], bpy0);
    atomicAdd(&ws[BSUM_OFF + b0 * K_SUP + col0], bs0);
    atomicAdd(&ws[BPOS_OFF + (b0 * K_SUP + col1) * 2 + 0], bpx1);
    atomicAdd(&ws[BPOS_OFF + (b0 * K_SUP + col1) * 2 + 1], bpy1);
    atomicAdd(&ws[BSUM_OFF + b0 * K_SUP + col1], bs1);
  }
  __syncthreads();

  // ---- pooled features: T[32,128] = s_blockT(32x64) @ x_block(64x128) via WMMA ----
  #pragma unroll
  for (int i = 0; i < 4; ++i) {
    int t = wave * 4 + i;
    int mtile = t >> 3, ntile = t & 7;
    v8f c = {};
    #pragma unroll
    for (int kk2 = 0; kk2 < 2; ++kk2) {
      const _Float16* ap = smkT + (mtile * 16 + l15) * 64 + kk2 * 32 + hi * 8;
      v16h aa = cat16(*(const v8h*)ap, *(const v8h*)(ap + 16));
      const _Float16* bp = xhT + (ntile * 16 + l15) * 64 + kk2 * 32 + hi * 8;
      v16h bb = cat16(*(const v8h*)bp, *(const v8h*)(bp + 16));
      c = __builtin_amdgcn_wmma_f32_16x16x32_f16(false, aa, false, bb,
                                                 (short)0, c, false, false);
    }
    #pragma unroll
    for (int v = 0; v < 8; ++v) {
      int ksup = mtile * 16 + v + 8 * hi;
      int ch = ntile * 16 + l15;
      atomicAdd(&d_pool[(b0 * K_SUP + ksup) * C_DIM + ch], c[v]);
    }
  }

  // ---- fallback for rows whose graph != b0 (graph-boundary blocks only) ----
  for (int lr = 0; lr < 64; ++lr) {
    int bb = batL[lr];
    if (bb != b0 && bb >= 0) {
      for (int e = tid; e < K_SUP * C_DIM; e += 128) {
        int k = e >> 7, ch = e & 127;
        atomicAdd(&d_pool[(bb * K_SUP + k) * C_DIM + ch],
                  sraw[lr * K_SUP + k] * (float)xhT[ch * 64 + lr]);
      }
    }
  }
}

__global__ __launch_bounds__(256) void shp_finalize_kernel(
    const float* __restrict__ ws, const float* __restrict__ amask,
    float* __restrict__ d_mu, float* __restrict__ d_losses) {
  __shared__ float mus[B_GRAPHS * K_SUP * 2];
  __shared__ float red[256];
  int tid = threadIdx.x;
  for (int i = tid; i < B_GRAPHS * K_SUP; i += 256) {
    float den = ws[BSUM_OFF + i] + EPSV;
    float mx = ws[BPOS_OFF + i * 2 + 0] / den;
    float my = ws[BPOS_OFF + i * 2 + 1] / den;
    mus[i * 2 + 0] = mx; mus[i * 2 + 1] = my;
    d_mu[i * 2 + 0] = mx; d_mu[i * 2 + 1] = my;
  }
  __syncthreads();
  float sep = 0.0f;
  for (int p = tid; p < B_GRAPHS * K_SUP * K_SUP; p += 256) {
    int b = p / (K_SUP * K_SUP);
    int ij = p - b * K_SUP * K_SUP;
    int i = ij >> 5, j = ij & 31;
    if (i != j) {
      float dx = mus[(b * K_SUP + i) * 2 + 0] - mus[(b * K_SUP + j) * 2 + 0];
      float dy = mus[(b * K_SUP + i) * 2 + 1] - mus[(b * K_SUP + j) * 2 + 1];
      sep += 1.0f / (dx * dx + dy * dy + 1.0f);
    }
  }
  red[tid] = sep;
  __syncthreads();
  for (int s = 128; s > 0; s >>= 1) {
    if (tid < s) red[tid] += red[tid + s];
    __syncthreads();
  }
  if (tid == 0) {
    float separation = red[0] / ((float)(K_SUP * (K_SUP - 1)) + EPSV);
    float invN = 1.0f / (float)N_NODES;
    float avg[K_SUP];
    float sum_avg = 0.0f, maxavg = -1e30f;
    for (int k = 0; k < K_SUP; ++k) {
      avg[k] = ws[AVG_OFF + k] * invN;
      sum_avg += avg[k];
      maxavg = fmaxf(maxavg, avg[k]);
    }
    float entropy = -ws[ENT_OFF] * invN;
    float diversity = 0, ent_avg = 0, balance = 0, pruning = 0, spatial = 0;
    const float u = 1.0f / (float)K_SUP;
    for (int k = 0; k < K_SUP; ++k) {
      float l = __logf(avg[k] + EPSV);
      diversity += avg[k] * l;
      ent_avg -= avg[k] * l;
      balance += u * __logf(u / (avg[k] + EPSV));
      pruning += fabsf(avg[k] * (1.0f - amask[k]));
      float den = ws[AVG_OFF + k] + EPSV;
      float mux = ws[MU_OFF + k * 2 + 0] / den;
      float muy = ws[MU_OFF + k * 2 + 1] / den;
      float Pk = ws[P_OFF + k] / den;
      spatial += Pk - (mux * mux + muy * muy);
    }
    pruning *= u;
    spatial *= u;
    float msum = 0.0f;
    for (int k = 0; k < K_SUP; ++k) msum += amask[k];
    float sparsity = msum * u * 0.01f;
    float mean = sum_avg * u;
    float var = 0.0f;
    for (int k = 0; k < K_SUP; ++k) { float d = avg[k] - mean; var += d * d; }
    var /= (float)(K_SUP - 1);
    float mp = fmaxf(maxavg - 0.8f, 0.0f); mp = mp * mp * 10.0f;
    float ep = fmaxf(0.5f * __logf((float)K_SUP) - ent_avg, 0.0f); ep = ep * ep;
    float collapse = (var + mp + ep) * 2.0f;
    d_losses[0] = entropy;   d_losses[1] = diversity; d_losses[2] = spatial;
    d_losses[3] = pruning;   d_losses[4] = sparsity;  d_losses[5] = 0.0f;
    d_losses[6] = collapse;  d_losses[7] = balance;   d_losses[8] = separation;
  }
}

extern "C" void kernel_launch(void* const* d_in, const int* in_sizes, int n_in,
                              void* d_out, int out_size, void* d_ws, size_t ws_size,
                              hipStream_t stream) {
  const float* x       = (const float*)d_in[0];
  const float* pos     = (const float*)d_in[1];
  const float* W1      = (const float*)d_in[2];
  const float* b1      = (const float*)d_in[3];
  const float* W2      = (const float*)d_in[4];
  const float* b2      = (const float*)d_in[5];
  const float* scaling = (const float*)d_in[6];
  const float* amask   = (const float*)d_in[7];
  const int*   batch   = (const int*)d_in[8];
  float* out = (float*)d_out;
  float* ws  = (float*)d_ws;

  shp_zero_kernel<<<128, 256, 0, stream>>>(out, ws);
  int nblocks = (N_NODES + 63) / 64;
  shp_main_kernel<<<nblocks, 128, LDS_TOTAL, stream>>>(
      x, pos, W1, b1, W2, b2, scaling, amask, batch,
      out, out + S_OFF, ws);
  shp_finalize_kernel<<<1, 256, 0, stream>>>(ws, amask, out + MUOUT_OFF, out + LOSS_OFF);
}
